// GreedyInference_33397665694101
// MI455X (gfx1250) — matrix operand order repaired and checked by reading, not compiled, not run on hardware
//
#include <hip/hip_runtime.h>
#include <hip/hip_bf16.h>
#include <math.h>

#define B_    32
#define T_    100
#define E_    512
#define H_    640
#define H4_   2560
#define J_    640
#define VP1_  4097
#define NPAD_ 4112      // 257*16, padded V+1
#define BLANK_ 4096
#define MS_   2

typedef __attribute__((ext_vector_type(16))) _Float16 v16h;
typedef __attribute__((ext_vector_type(8)))  float    v8f;
typedef int rnnt_v4i __attribute__((__vector_size__(16)));

// A-fragment (16x32 f16): lane holds row M=lane%16; K = 8*(lane/16) + {0..7, 16..23}
__device__ __forceinline__ v16h rnnt_load_a(const _Float16* rowptr, int lanehi) {
  v16h r;
  ((uint4*)&r)[0] = *(const uint4*)(rowptr + 8 * lanehi);
  ((uint4*)&r)[1] = *(const uint4*)(rowptr + 8 * lanehi + 16);
  return r;
}
// B-fragment (32x16 f16): lane holds col N=lane%16; K = 8*(lane/16) + {0..7, 16..23}
__device__ __forceinline__ v16h rnnt_load_b(const _Float16* rowptr, int lanehi) {
  v16h r;
  ((uint4*)&r)[0] = *(const uint4*)(rowptr + 8 * lanehi);
  ((uint4*)&r)[1] = *(const uint4*)(rowptr + 8 * lanehi + 16);
  return r;
}
#define WMMA_F16(acc, a, b) \
  (acc) = __builtin_amdgcn_wmma_f32_16x16x32_f16(false, (a), false, (b), (short)0, (acc), false, false)

// ---- CDNA5 async global->LDS copy (16B per lane), tracked by ASYNCcnt ----
__device__ __forceinline__ void rnnt_g2l_b128(void* lds, const void* gptr) {
#if __has_builtin(__builtin_amdgcn_global_load_async_to_lds_b128)
  __builtin_amdgcn_global_load_async_to_lds_b128(
      (rnnt_v4i*)gptr, (rnnt_v4i*)lds, 0, 0);
#else
  unsigned loff = (unsigned)(unsigned long long)lds;   // low 32 bits = LDS offset
  asm volatile("global_load_async_to_lds_b128 %0, %1, off"
               :: "v"(loff), "v"(gptr) : "memory");
#endif
}
__device__ __forceinline__ void rnnt_wait_async0() {
#if __has_builtin(__builtin_amdgcn_s_wait_asynccnt)
  __builtin_amdgcn_s_wait_asynccnt(0);
#else
  asm volatile("s_wait_asynccnt 0x0" ::: "memory");
#endif
}

// ---------- prep kernels ----------

// dst[n*K + k] = (n < N) ? src[k*N + n] : 0   (f32 -> f16 transpose with N-pad)
__global__ void rnnt_tr_f32f16(const float* __restrict__ src, _Float16* __restrict__ dst,
                               int K, int N, int Npad) {
  int i = blockIdx.x * blockDim.x + threadIdx.x;
  int tot = K * Npad;
  if (i >= tot) return;
  int n = i / K, k = i % K;
  dst[i] = (n < N) ? (_Float16)src[(size_t)k * N + n] : (_Float16)0.f;
}

__global__ void rnnt_f32tof16(const float* __restrict__ s, _Float16* __restrict__ d, int n) {
  int i = blockIdx.x * blockDim.x + threadIdx.x;
  if (i < n) d[i] = (_Float16)s[i];
}

__global__ void rnnt_biaspad(const float* __restrict__ s, float* __restrict__ d) {
  int i = blockIdx.x * blockDim.x + threadIdx.x;
  if (i < NPAD_) d[i] = (i < VP1_) ? s[i] : -1.0e30f;
}

// enc [B,E,T] f32  ->  enc16 [B*T, E] f16 (row-major over e)
__global__ void rnnt_enc_tr(const float* __restrict__ enc, _Float16* __restrict__ dst) {
  int i = blockIdx.x * blockDim.x + threadIdx.x;   // over [b][e][t] (source-contiguous)
  if (i >= B_ * E_ * T_) return;
  int t = i % T_;
  int e = (i / T_) % E_;
  int b = i / (E_ * T_);
  dst[((size_t)(b * T_ + t)) * E_ + e] = (_Float16)enc[i];
}

// ---------- enc_proj = enc16 @ W_enc^T  via WMMA ----------
__global__ void rnnt_encproj(const _Float16* __restrict__ enc16, const _Float16* __restrict__ WencT,
                             float* __restrict__ encp) {
  int gwave = (blockIdx.x * blockDim.x + threadIdx.x) >> 5;
  int lane  = threadIdx.x & 31;
  const int MT = (B_ * T_) / 16, NT = J_ / 16;     // 200 x 40 tiles
  if (gwave >= MT * NT) return;
  int mt = gwave / NT, nt = gwave % NT;
  int lanelo = lane & 15, lanehi = lane >> 4;

  const _Float16* arow = enc16 + (size_t)(mt * 16 + lanelo) * E_;
  const _Float16* brow = WencT + (size_t)(nt * 16 + lanelo) * E_;

  v8f acc = {};
  for (int kk = 0; kk < E_; kk += 32) {
    v16h a = rnnt_load_a(arow + kk, lanehi);
    v16h b = rnnt_load_b(brow + kk, lanehi);
    WMMA_F16(acc, a, b);
  }
#pragma unroll
  for (int v = 0; v < 8; ++v) {
    int md = mt * 16 + v + 8 * lanehi;
    encp[(size_t)md * J_ + nt * 16 + lanelo] = acc[v];
  }
}

// ---------- persistent sequential decoder: 1 block, 1024 threads (32 waves) ----------
__global__ __launch_bounds__(1024) void rnnt_decode(
    const int* __restrict__ lengths, const float* __restrict__ b_lstm,
    const float* __restrict__ b_joint,
    const _Float16* __restrict__ WiT, const _Float16* __restrict__ WhT,
    const _Float16* __restrict__ WpT, const _Float16* __restrict__ WoT,
    const _Float16* __restrict__ emb16, const float* __restrict__ boutp,
    const float* __restrict__ encp,
    float* __restrict__ logitsW, float* __restrict__ cnewW, int* __restrict__ labelsW,
    float* __restrict__ out) {
  extern __shared__ char smem[];
  float*    cS  = (float*)smem;                                    // [32][640] f32
  _Float16* hS  = (_Float16*)(smem + 81920);                       // [32][640] f16
  _Float16* hnS = (_Float16*)(smem + 81920 + 40960);               // h1 candidate
  _Float16* xS  = (_Float16*)(smem + 81920 + 81920);               // embed input
  _Float16* rS  = (_Float16*)(smem + 81920 + 122880);              // relu(joint hidden)
  char* st = smem + 81920 + 163840;
  int*   lastS = (int*)(st);
  int*   cntS  = (int*)(st + 128);
  int*   blkS  = (int*)(st + 256);
  int*   cmtS  = (int*)(st + 384);
  float* scoS  = (float*)(st + 512);

  const int tid = threadIdx.x, lane = tid & 31, wave = tid >> 5;
  const int lanelo = lane & 15, lanehi = lane >> 4;

  for (int i = tid; i < B_ * H_; i += 1024) { cS[i] = 0.f; hS[i] = (_Float16)0.f; }
  for (int i = tid; i < B_ * MS_ * T_; i += 1024) labelsW[i] = BLANK_;
  if (tid < B_) { lastS[tid] = BLANK_; cntS[tid] = 0; scoS[tid] = 0.f; }
  __syncthreads();

  for (int t = 0; t < T_; ++t) {
    if (tid < B_) blkS[tid] = (t >= lengths[tid]) ? 1 : 0;
    __syncthreads();
    for (int s = 0; s < MS_; ++s) {
      const bool sos = (t == 0 && s == 0);
      // ---- x = sos ? 0 : embed[last_label]  (async global->LDS gather) ----
      {
        const int CH = (B_ * H_) / 8;                // 16B chunks of xS
        if (sos) {
          uint4 z = {0u, 0u, 0u, 0u};
          for (int i = tid; i < CH; i += 1024) ((uint4*)xS)[i] = z;
        } else {
          for (int i = tid; i < CH; i += 1024) {
            int b = i / (H_ / 8);
            int k = (i % (H_ / 8)) * 8;
            const _Float16* gp = emb16 + (size_t)lastS[b] * H_ + k;
            rnnt_g2l_b128(xS + (size_t)i * 8, gp);
          }
          rnnt_wait_async0();
        }
      }
      __syncthreads();

      // ---- LSTM: z = x@Wi + h@Wh + b; 80 (mtile, hidden-blk) units ----
      for (int u = wave; u < 2 * (H_ / 16); u += 32) {
        int mt = u & 1, nb = u >> 1;
        int hidx = nb * 16 + lanelo;
        v8f ai = {}, af = {}, ag = {}, ao = {};
        const _Float16* bi0 = WiT + (size_t)(0 * H_ + hidx) * H_;
        const _Float16* bi1 = WiT + (size_t)(1 * H_ + hidx) * H_;
        const _Float16* bi2 = WiT + (size_t)(2 * H_ + hidx) * H_;
        const _Float16* bi3 = WiT + (size_t)(3 * H_ + hidx) * H_;
        const _Float16* arow = xS + (size_t)(mt * 16 + lanelo) * H_;
        for (int kk = 0; kk < H_; kk += 32) {
          v16h a = rnnt_load_a(arow + kk, lanehi);
          WMMA_F16(ai, a, rnnt_load_b(bi0 + kk, lanehi));
          WMMA_F16(af, a, rnnt_load_b(bi1 + kk, lanehi));
          WMMA_F16(ag, a, rnnt_load_b(bi2 + kk, lanehi));
          WMMA_F16(ao, a, rnnt_load_b(bi3 + kk, lanehi));
        }
        const _Float16* bh0 = WhT + (size_t)(0 * H_ + hidx) * H_;
        const _Float16* bh1 = WhT + (size_t)(1 * H_ + hidx) * H_;
        const _Float16* bh2 = WhT + (size_t)(2 * H_ + hidx) * H_;
        const _Float16* bh3 = WhT + (size_t)(3 * H_ + hidx) * H_;
        arow = hS + (size_t)(mt * 16 + lanelo) * H_;
        for (int kk = 0; kk < H_; kk += 32) {
          v16h a = rnnt_load_a(arow + kk, lanehi);
          WMMA_F16(ai, a, rnnt_load_b(bh0 + kk, lanehi));
          WMMA_F16(af, a, rnnt_load_b(bh1 + kk, lanehi));
          WMMA_F16(ag, a, rnnt_load_b(bh2 + kk, lanehi));
          WMMA_F16(ao, a, rnnt_load_b(bh3 + kk, lanehi));
        }
        float bbi = b_lstm[0 * H_ + hidx], bbf = b_lstm[1 * H_ + hidx];
        float bbg = b_lstm[2 * H_ + hidx], bbo = b_lstm[3 * H_ + hidx];
#pragma unroll
        for (int v = 0; v < 8; ++v) {
          int m = mt * 16 + v + 8 * lanehi;
          float iv = 1.f / (1.f + __expf(-(ai[v] + bbi)));
          float fv = 1.f / (1.f + __expf(-(af[v] + bbf)));
          float gv = tanhf(ag[v] + bbg);
          float ov = 1.f / (1.f + __expf(-(ao[v] + bbo)));
          float cn = fv * cS[m * H_ + hidx] + iv * gv;
          float hn = ov * tanhf(cn);
          cnewW[m * H_ + hidx] = cn;
          hnS[m * H_ + hidx]   = (_Float16)hn;
        }
      }
      __syncthreads();

      // ---- joint hidden: r = relu(enc_proj[:,t] + h1@Wp + b_joint) ----
      for (int u = wave; u < 2 * (J_ / 16); u += 32) {
        int mt = u & 1, nb = u >> 1;
        int jc = nb * 16 + lanelo;
        v8f acc = {};
        const _Float16* arow = hnS + (size_t)(mt * 16 + lanelo) * H_;
        const _Float16* brow = WpT + (size_t)jc * H_;
        for (int kk = 0; kk < H_; kk += 32) {
          v16h a = rnnt_load_a(arow + kk, lanehi);
          WMMA_F16(acc, a, rnnt_load_b(brow + kk, lanehi));
        }
        float bj = b_joint[jc];
#pragma unroll
        for (int v = 0; v < 8; ++v) {
          int m = mt * 16 + v + 8 * lanehi;
          float val = acc[v] + encp[((size_t)m * T_ + t) * J_ + jc] + bj;
          rS[m * H_ + jc] = (_Float16)(val > 0.f ? val : 0.f);
        }
      }
      __syncthreads();

      // ---- logits = r @ W_out + b_out (N padded to 4112) ----
      for (int u = wave; u < 2 * (NPAD_ / 16); u += 32) {
        int mt = u & 1, nb = u >> 1;
        int nc = nb * 16 + lanelo;
        v8f acc = {};
        const _Float16* arow = rS + (size_t)(mt * 16 + lanelo) * H_;
        const _Float16* brow = WoT + (size_t)nc * H_;
        for (int kk = 0; kk < H_; kk += 32) {
          __builtin_prefetch(brow + kk + 256, 0, 1);
          v16h a = rnnt_load_a(arow + kk, lanehi);
          WMMA_F16(acc, a, rnnt_load_b(brow + kk, lanehi));
        }
        float bb = boutp[nc];
#pragma unroll
        for (int v = 0; v < 8; ++v) {
          int m = mt * 16 + v + 8 * lanehi;
          logitsW[(size_t)m * NPAD_ + nc] = acc[v] + bb;
        }
      }
      __threadfence_block();
      __syncthreads();

      // ---- per-row argmax + logsumexp + commit: wave owns batch row b=wave ----
      {
        int b = wave;
        const float* row = logitsW + (size_t)b * NPAD_;
        float mv = -3.0e38f; int mi = NPAD_;
        for (int c = lane; c < NPAD_; c += 32) {
          float v = row[c];
          if (v > mv || (v == mv && c < mi)) { mv = v; mi = c; }
        }
        for (int off = 16; off > 0; off >>= 1) {
          float ov = __shfl_xor(mv, off, 32);
          int   oi = __shfl_xor(mi, off, 32);
          if (ov > mv || (ov == mv && oi < mi)) { mv = ov; mi = oi; }
        }
        float sum = 0.f;
        for (int c = lane; c < NPAD_; c += 32) sum += __expf(row[c] - mv);
        for (int off = 16; off > 0; off >>= 1) sum += __shfl_xor(sum, off, 32);
        if (lane == 0) {
          float vlp = -__logf(sum);                 // logp at argmax
          int nbk = blkS[b] | ((mi == BLANK_) ? 1 : 0);
          int cm  = nbk ? 0 : 1;
          if (cm) {
            lastS[b] = mi;
            labelsW[b * (MS_ * T_) + cntS[b]] = mi;
            cntS[b] += 1;
            scoS[b] += vlp;
          }
          blkS[b] = nbk;
          cmtS[b] = cm;
        }
      }
      __syncthreads();

      // ---- conditionally commit h (LDS->LDS) and c (async global->LDS) ----
      for (int i = tid; i < (B_ * H_) / 4; i += 1024) {     // 16B chunks of cS (f32)
        int b = i / (H_ / 4);
        if (cmtS[b]) rnnt_g2l_b128((char*)cS + (size_t)i * 16,
                                   (const char*)cnewW + (size_t)i * 16);
      }
      for (int i = tid; i < (B_ * H_) / 8; i += 1024) {     // 16B chunks of hS (f16)
        int b = i / (H_ / 8);
        if (cmtS[b]) ((uint4*)hS)[i] = ((const uint4*)hnS)[i];
      }
      rnnt_wait_async0();
      __syncthreads();
    }
  }

  // outputs: labels (as float), counts, score
  for (int i = tid; i < B_ * MS_ * T_; i += 1024) out[i] = (float)labelsW[i];
  if (tid < B_) {
    out[B_ * MS_ * T_ + tid]      = (float)cntS[tid];
    out[B_ * MS_ * T_ + B_ + tid] = scoS[tid];
  }
}

extern "C" void kernel_launch(void* const* d_in, const int* in_sizes, int n_in,
                              void* d_out, int out_size, void* d_ws, size_t ws_size,
                              hipStream_t stream) {
  const float* enc     = (const float*)d_in[0];
  const int*   lens    = (const int*)d_in[1];
  const float* embed   = (const float*)d_in[2];
  const float* W_i     = (const float*)d_in[3];
  const float* W_h     = (const float*)d_in[4];
  const float* b_lstm  = (const float*)d_in[5];
  const float* W_enc   = (const float*)d_in[6];
  const float* W_pred  = (const float*)d_in[7];
  const float* b_joint = (const float*)d_in[8];
  const float* W_out   = (const float*)d_in[9];
  const float* b_out   = (const float*)d_in[10];

  char* ws = (char*)d_ws;
  size_t off = 0;
  auto alloc = [&](size_t bytes) { size_t r = off; off += (bytes + 255) & ~(size_t)255; return r; };
  _Float16* WiT   = (_Float16*)(ws + alloc((size_t)H4_ * H_ * 2));
  _Float16* WhT   = (_Float16*)(ws + alloc((size_t)H4_ * H_ * 2));
  _Float16* WpT   = (_Float16*)(ws + alloc((size_t)J_ * H_ * 2));
  _Float16* WoT   = (_Float16*)(ws + alloc((size_t)NPAD_ * H_ * 2));
  _Float16* WencT = (_Float16*)(ws + alloc((size_t)J_ * E_ * 2));
  _Float16* emb16 = (_Float16*)(ws + alloc((size_t)VP1_ * H_ * 2));
  _Float16* enc16 = (_Float16*)(ws + alloc((size_t)B_ * T_ * E_ * 2));
  float* boutp    = (float*)(ws + alloc((size_t)NPAD_ * 4));
  float* encp     = (float*)(ws + alloc((size_t)B_ * T_ * J_ * 4));
  float* logitsW  = (float*)(ws + alloc((size_t)B_ * NPAD_ * 4));
  float* cnewW    = (float*)(ws + alloc((size_t)B_ * H_ * 4));
  int*   labelsW  = (int*)(ws + alloc((size_t)B_ * MS_ * T_ * 4));

  const int thr = 256;
  rnnt_tr_f32f16<<<(H_ * H4_ + thr - 1) / thr, thr, 0, stream>>>(W_i, WiT, H_, H4_, H4_);
  rnnt_tr_f32f16<<<(H_ * H4_ + thr - 1) / thr, thr, 0, stream>>>(W_h, WhT, H_, H4_, H4_);
  rnnt_tr_f32f16<<<(H_ * J_ + thr - 1) / thr, thr, 0, stream>>>(W_pred, WpT, H_, J_, J_);
  rnnt_tr_f32f16<<<(H_ * NPAD_ + thr - 1) / thr, thr, 0, stream>>>(W_out, WoT, H_, VP1_, NPAD_);
  rnnt_tr_f32f16<<<(E_ * J_ + thr - 1) / thr, thr, 0, stream>>>(W_enc, WencT, E_, J_, J_);
  rnnt_f32tof16<<<(VP1_ * H_ + thr - 1) / thr, thr, 0, stream>>>(embed, emb16, VP1_ * H_);
  rnnt_enc_tr<<<(B_ * E_ * T_ + thr - 1) / thr, thr, 0, stream>>>(enc, enc16);
  rnnt_biaspad<<<(NPAD_ + thr - 1) / thr, thr, 0, stream>>>(b_out, boutp);

  const int nTiles = ((B_ * T_) / 16) * (J_ / 16);   // 8000 waves
  rnnt_encproj<<<(nTiles * 32 + thr - 1) / thr, thr, 0, stream>>>(enc16, WencT, encp);

  size_t smem = 81920 /*c*/ + 4 * 40960 /*h,hn,x,r*/ + 1024 /*state*/;
  rnnt_decode<<<1, 1024, smem, stream>>>(lens, b_lstm, b_joint, WiT, WhT, WpT, WoT,
                                         emb16, boutp, encp, logitsW, cnewW, labelsW,
                                         (float*)d_out);
}
